// MTHiSAN_46188078301259
// MI455X (gfx1250) — matrix operand order repaired and checked by reading, not compiled, not run on hardware
//
#include <hip/hip_runtime.h>

// ---------------------------------------------------------------------------
// MTHiSAN forward for MI455X (gfx1250): WMMA bf16 GEMMs + VALU attention.
// ---------------------------------------------------------------------------

typedef __attribute__((ext_vector_type(16))) __bf16 v16bf;
typedef __attribute__((ext_vector_type(8)))  float  v8f;

union V16 { v16bf v; unsigned short u[16]; };

__device__ __forceinline__ unsigned short f32_to_bf16(float f) {
  union { float f; unsigned u; } x; x.f = f;
  unsigned u = x.u;
  unsigned r = (u + 0x7FFFu + ((u >> 16) & 1u)) >> 16;   // round-to-nearest-even
  return (unsigned short)r;
}
__device__ __forceinline__ float bf16_to_f32(unsigned short h) {
  union { unsigned u; float f; } x; x.u = ((unsigned)h) << 16;
  return x.f;
}

// ---------------------------------------------------------------------------
// Kernel 1/3: C[M,1200] = elu( A[M,K] @ [Wq|Wk|Wv] + bias ), bf16 output.
// A is either gathered rows of `emb` via `ids` (word level) or dense (line
// level).  One WG = 64 rows x 128 cols; 8 waves, each wave owns one 16-wide
// n-tile and runs 4 WMMAs (4 m-tiles) per k-step sharing one B fragment.
// ---------------------------------------------------------------------------
#define LDS_STRIDE 48   // ushorts; 96B row pitch: 16B-aligned, bank-friendly

__global__ __launch_bounds__(256)
void qkv_gemm(const float* __restrict__ Adense,
              const int*   __restrict__ ids,
              const float* __restrict__ Agather,
              int M, int K,
              const float* __restrict__ Wq, const float* __restrict__ Wk,
              const float* __restrict__ Wv,
              const float* __restrict__ bq, const float* __restrict__ bk,
              const float* __restrict__ bv,
              unsigned short* __restrict__ out) {
  __shared__ __align__(16) unsigned short As[64 * LDS_STRIDE];   // A: 64 rows x 32 k (bf16)
  __shared__ __align__(16) unsigned short Bt[128 * LDS_STRIDE];  // B^T: 128 cols x 32 k (bf16)

  const int mtile  = blockIdx.x;             // 64-row block
  const int ngroup = blockIdx.y;             // 8 n-tiles of 16 per group
  const int tid    = threadIdx.x;            // 256 threads = 8 waves
  const int wave   = tid >> 5;
  const int lane   = tid & 31;
  const int ntile  = ngroup * 8 + wave;      // valid if < 75
  const int nbase  = ntile * 16;
  const int rowBase = mtile * 64;
  const int ksteps = (K + 31) / 32;

  v8f acc0 = {0.f,0.f,0.f,0.f,0.f,0.f,0.f,0.f};
  v8f acc1 = acc0, acc2 = acc0, acc3 = acc0;

  for (int ks = 0; ks < ksteps; ++ks) {
    const int k0 = ks * 32;
    // stage A tile (64x32), row-contiguous per lane
    for (int e = tid; e < 64 * 32; e += 256) {
      int r = e >> 5, c = e & 31;
      int gr = rowBase + r, gk = k0 + c;
      float v = 0.f;
      if (gr < M && gk < K) {
        if (ids) { int id = ids[gr]; v = Agather[(long long)id * K + gk]; }
        else     { v = Adense[(long long)gr * K + gk]; }
      }
      As[r * LDS_STRIDE + c] = f32_to_bf16(v);
    }
    // stage B slab transposed: Bt[col][k] so lane fragments are contiguous
    const int colBase = ngroup * 128;
    for (int e = tid; e < 32 * 128; e += 256) {
      int kr = e >> 7, c = e & 127;
      int gk = k0 + kr, gn = colBase + c;
      float v = 0.f;
      if (gk < K && gn < 1200) {
        int sel = gn / 400, c2 = gn % 400;
        const float* W = (sel == 0) ? Wq : ((sel == 1) ? Wk : Wv);
        v = W[(long long)gk * 400 + c2];
      }
      Bt[c * LDS_STRIDE + kr] = f32_to_bf16(v);
    }
    __syncthreads();

    if (ntile < 75) {
      const int hi = lane >> 4;      // half-wave
      const int lr = lane & 15;      // A: row M ; B/C: col N
      // B layout: lanes 0-15 K=0..7 & 16..23 ; lanes 16-31 K=8..15 & 24..31, col N=lr
      V16 b;
      const unsigned short* bp = &Bt[(wave * 16 + lr) * LDS_STRIDE + hi * 8];
      #pragma unroll
      for (int i = 0; i < 8; ++i) { b.u[i] = bp[i]; b.u[8 + i] = bp[16 + i]; }
      // A layout: lanes 0-15 K=0..7 & 16..23 ; lanes 16-31 K=8..15 & 24..31
      #pragma unroll
      for (int mt = 0; mt < 4; ++mt) {
        V16 a;
        const unsigned short* ap = &As[(mt * 16 + lr) * LDS_STRIDE + hi * 8];
        #pragma unroll
        for (int i = 0; i < 8; ++i) a.u[i]     = ap[i];
        #pragma unroll
        for (int i = 0; i < 8; ++i) a.u[8 + i] = ap[16 + i];
        v8f& acc = (mt == 0) ? acc0 : (mt == 1) ? acc1 : (mt == 2) ? acc2 : acc3;
        acc = __builtin_amdgcn_wmma_f32_16x16x32_bf16(
            false, a.v, false, b.v, (short)0, acc, false, false);
      }
    }
    __syncthreads();
  }

  if (ntile < 75) {
    const int hi = lane >> 4;
    const int n  = lane & 15;
    const int gn = nbase + n;
    const int sel = gn / 400, c2 = gn % 400;
    const float* bb = (sel == 0) ? bq : ((sel == 1) ? bk : bv);
    const float bias = bb[c2];
    #pragma unroll
    for (int mt = 0; mt < 4; ++mt) {
      v8f& acc = (mt == 0) ? acc0 : (mt == 1) ? acc1 : (mt == 2) ? acc2 : acc3;
      #pragma unroll
      for (int r = 0; r < 8; ++r) {
        int gm = rowBase + mt * 16 + r + hi * 8;   // C layout: M = r + 8*hi
        if (gm < M) {
          float x = acc[r] + bias;
          x = (x > 0.f) ? x : (__expf(x) - 1.f);   // ELU
          out[(long long)gm * 1200 + gn] = f32_to_bf16(x);
        }
      }
    }
  }
}

// ---------------------------------------------------------------------------
// Kernel 2: word-level attention + target attention.  WG = (line, head).
// ---------------------------------------------------------------------------
__global__ void word_attn(const unsigned short* __restrict__ qkv,
                          const int* __restrict__ docs,
                          const float* __restrict__ w_tgt,
                          float* __restrict__ le) {
  const int line = blockIdx.x;   // 0..6399
  const int h    = blockIdx.y;   // 0..7
  const int tid  = threadIdx.x;  // 64 threads

  __shared__ float Q[30][50], Kh[30][50], Vh[30][50];
  __shared__ float S[30][30];
  __shared__ float Wt[30][50];
  __shared__ float m[30];
  __shared__ float ts[32];
  __shared__ float tnorm;

  for (int e = tid; e < 30 * 50; e += 64) {
    int w = e / 50, d = e % 50;
    long long base = (long long)(line * 30 + w) * 1200;
    Q[w][d]  = bf16_to_f32(qkv[base +        h * 50 + d]);
    Kh[w][d] = bf16_to_f32(qkv[base + 400 +  h * 50 + d]);
    Vh[w][d] = bf16_to_f32(qkv[base + 800 +  h * 50 + d]);
  }
  for (int w = tid; w < 30; w += 64) m[w] = (docs[line * 30 + w] != 0) ? 1.f : 0.f;
  __syncthreads();

  const float scale = 0.14142135623730951f;  // 1/sqrt(50)
  for (int e = tid; e < 900; e += 64) {
    int i = e / 30, j = e % 30;
    float s = 0.f;
    for (int d = 0; d < 50; ++d) s += Q[i][d] * Kh[j][d];
    S[i][j] = s * scale - 1e7f * (1.f - m[j]);
  }
  __syncthreads();
  for (int i = tid; i < 30; i += 64) {
    float mx = -1e30f;
    for (int j = 0; j < 30; ++j) mx = fmaxf(mx, S[i][j]);
    float sum = 0.f;
    for (int j = 0; j < 30; ++j) { float e2 = __expf(S[i][j] - mx); S[i][j] = e2; sum += e2; }
    float inv = m[i] / sum;                 // query mask folded into weights
    for (int j = 0; j < 30; ++j) S[i][j] *= inv;
  }
  __syncthreads();
  for (int e = tid; e < 1500; e += 64) {
    int i = e / 50, d = e % 50;
    float r = 0.f;
    for (int j = 0; j < 30; ++j) r += S[i][j] * Vh[j][d];
    Wt[i][d] = r;                           // masked rows already zero
  }
  __syncthreads();
  for (int j = tid; j < 30; j += 64) {
    float s = 0.f;
    for (int d = 0; d < 50; ++d) s += w_tgt[h * 50 + d] * Wt[j][d];
    ts[j] = s * scale - 1e7f * (1.f - m[j]);
  }
  __syncthreads();
  if (tid == 0) {
    float mx = -1e30f;
    for (int j = 0; j < 30; ++j) mx = fmaxf(mx, ts[j]);
    float sum = 0.f;
    for (int j = 0; j < 30; ++j) { float e2 = __expf(ts[j] - mx); ts[j] = e2; sum += e2; }
    tnorm = sum;
  }
  __syncthreads();
  float lm = 0.f;
  for (int w = 0; w < 30; ++w) lm = fmaxf(lm, m[w]);   // line non-empty mask
  const float inv = lm / tnorm;
  for (int d = tid; d < 50; d += 64) {
    float r = 0.f;
    for (int j = 0; j < 30; ++j) r += ts[j] * Wt[j][d];
    le[(long long)line * 400 + h * 50 + d] = r * inv;
  }
}

// ---------------------------------------------------------------------------
// Kernel 4: line-level attention + target attention.  WG = (doc, head).
// ---------------------------------------------------------------------------
__global__ void line_attn(const unsigned short* __restrict__ lqkv,
                          const int* __restrict__ docs,
                          const float* __restrict__ l_tgt,
                          float* __restrict__ de) {
  const int b   = blockIdx.x;   // 0..63
  const int h   = blockIdx.y;   // 0..7
  const int tid = threadIdx.x;  // 128 threads

  __shared__ float Kh[100][50];
  __shared__ float Vh[100][50];
  __shared__ float La[100][50];
  __shared__ float ml[100];
  __shared__ float ts[128];
  __shared__ float tnorm;

  for (int e = tid; e < 100 * 50; e += 128) {
    int l = e / 50, d = e % 50;
    long long base = (long long)(b * 100 + l) * 1200;
    Kh[l][d] = bf16_to_f32(lqkv[base + 400 + h * 50 + d]);
    Vh[l][d] = bf16_to_f32(lqkv[base + 800 + h * 50 + d]);
  }
  for (int l = tid; l < 100; l += 128) {
    int any = 0;
    const int* p = docs + (long long)b * 3000 + l * 30;
    for (int w = 0; w < 30; ++w) any |= (p[w] != 0);
    ml[l] = any ? 1.f : 0.f;
  }
  __syncthreads();

  const float scale = 0.14142135623730951f;
  if (tid < 100) {
    const int l = tid;
    float q[50];
    long long base = (long long)(b * 100 + l) * 1200;
    for (int d = 0; d < 50; ++d) q[d] = bf16_to_f32(lqkv[base + h * 50 + d]);
    float s[100];
    float mx = -1e30f;
    for (int j = 0; j < 100; ++j) {
      float acc = 0.f;
      for (int d = 0; d < 50; ++d) acc += q[d] * Kh[j][d];
      acc = acc * scale - 1e7f * (1.f - ml[j]);
      s[j] = acc;
      mx = fmaxf(mx, acc);
    }
    float sum = 0.f;
    for (int j = 0; j < 100; ++j) { float e2 = __expf(s[j] - mx); s[j] = e2; sum += e2; }
    const float inv = ml[l] / sum;
    for (int d = 0; d < 50; ++d) {
      float r = 0.f;
      for (int j = 0; j < 100; ++j) r += s[j] * Vh[j][d];
      La[l][d] = r * inv;
    }
  }
  __syncthreads();
  for (int j = tid; j < 100; j += 128) {
    float s = 0.f;
    for (int d = 0; d < 50; ++d) s += l_tgt[h * 50 + d] * La[j][d];
    ts[j] = s * scale - 1e7f * (1.f - ml[j]);
  }
  __syncthreads();
  if (tid == 0) {
    float mx = -1e30f;
    for (int j = 0; j < 100; ++j) mx = fmaxf(mx, ts[j]);
    float sum = 0.f;
    for (int j = 0; j < 100; ++j) { float e2 = __expf(ts[j] - mx); ts[j] = e2; sum += e2; }
    tnorm = sum;
  }
  __syncthreads();
  const float inv = 1.f / tnorm;
  for (int d = tid; d < 50; d += 128) {
    float r = 0.f;
    for (int j = 0; j < 100; ++j) r += ts[j] * La[j][d];
    de[(long long)b * 400 + h * 50 + d] = r * inv;
  }
}

// ---------------------------------------------------------------------------
// Kernel 5: 5-head classifier, concatenated logits [64, 792].
// ---------------------------------------------------------------------------
__global__ void classifier(const float* __restrict__ de,
                           const float* __restrict__ w0, const float* __restrict__ w1,
                           const float* __restrict__ w2, const float* __restrict__ w3,
                           const float* __restrict__ w4,
                           const float* __restrict__ b0, const float* __restrict__ b1,
                           const float* __restrict__ b2, const float* __restrict__ b3,
                           const float* __restrict__ b4,
                           float* __restrict__ out) {
  int idx = blockIdx.x * blockDim.x + threadIdx.x;
  if (idx >= 64 * 792) return;
  int b = idx / 792, c = idx % 792;
  const float* W; const float* bb; int col, width;
  if (c < 25)        { W = w0; bb = b0; col = c;       width = 25;  }
  else if (c < 142)  { W = w1; bb = b1; col = c - 25;  width = 117; }
  else if (c < 149)  { W = w2; bb = b2; col = c - 142; width = 7;   }
  else if (c < 153)  { W = w3; bb = b3; col = c - 149; width = 4;   }
  else               { W = w4; bb = b4; col = c - 153; width = 639; }
  float s = bb[col];
  const float* dr = de + b * 400;
  for (int k = 0; k < 400; ++k) s += dr[k] * W[k * width + col];
  out[idx] = s;
}

// ---------------------------------------------------------------------------
extern "C" void kernel_launch(void* const* d_in, const int* in_sizes, int n_in,
                              void* d_out, int out_size, void* d_ws, size_t ws_size,
                              hipStream_t stream) {
  const int*   docs  = (const int*)d_in[0];
  const float* emb   = (const float*)d_in[1];
  const float* wq_w  = (const float*)d_in[2];
  const float* wq_b  = (const float*)d_in[3];
  const float* wk_w  = (const float*)d_in[4];
  const float* wk_b  = (const float*)d_in[5];
  const float* wv_w  = (const float*)d_in[6];
  const float* wv_b  = (const float*)d_in[7];
  const float* w_tgt = (const float*)d_in[8];
  const float* lq_w  = (const float*)d_in[9];
  const float* lq_b  = (const float*)d_in[10];
  const float* lk_w  = (const float*)d_in[11];
  const float* lk_b  = (const float*)d_in[12];
  const float* lv_w  = (const float*)d_in[13];
  const float* lv_b  = (const float*)d_in[14];
  const float* l_tgt = (const float*)d_in[15];
  const float* cw0 = (const float*)d_in[16];
  const float* cw1 = (const float*)d_in[17];
  const float* cw2 = (const float*)d_in[18];
  const float* cw3 = (const float*)d_in[19];
  const float* cw4 = (const float*)d_in[20];
  const float* cb0 = (const float*)d_in[21];
  const float* cb1 = (const float*)d_in[22];
  const float* cb2 = (const float*)d_in[23];
  const float* cb3 = (const float*)d_in[24];
  const float* cb4 = (const float*)d_in[25];

  // workspace layout
  char* ws = (char*)d_ws;
  unsigned short* qkv_w = (unsigned short*)ws;                         // 192000*1200*2 bytes
  size_t off = (size_t)192000 * 1200 * 2;
  float* le = (float*)(ws + off);                                      // 6400*400*4
  off += (size_t)6400 * 400 * 4;
  unsigned short* lqkv = (unsigned short*)(ws + off);                  // 6400*1200*2
  off += (size_t)6400 * 1200 * 2;
  float* de = (float*)(ws + off);                                      // 64*400*4

  // 1) word-level QKV projection (embedding gather + WMMA GEMM + ELU)
  qkv_gemm<<<dim3(3000, 10), 256, 0, stream>>>(
      nullptr, docs, emb, 192000, 300,
      wq_w, wk_w, wv_w, wq_b, wk_b, wv_b, qkv_w);

  // 2) word attention + target attention -> line embeddings
  word_attn<<<dim3(6400, 8), 64, 0, stream>>>(qkv_w, docs, w_tgt, le);

  // 3) line-level QKV projection (WMMA GEMM + ELU)
  qkv_gemm<<<dim3(100, 10), 256, 0, stream>>>(
      le, nullptr, nullptr, 6400, 400,
      lq_w, lk_w, lv_w, lq_b, lk_b, lv_b, lqkv);

  // 4) line attention + target attention -> doc embeddings
  line_attn<<<dim3(64, 8), 128, 0, stream>>>(lqkv, docs, l_tgt, de);

  // 5) classifiers
  classifier<<<dim3((64 * 792 + 255) / 256), 256, 0, stream>>>(
      de, cw0, cw1, cw2, cw3, cw4, cb0, cb1, cb2, cb3, cb4, (float*)d_out);
}
